// StyleExtractorV2_4243427689219
// MI455X (gfx1250) — matrix operand order repaired and checked by reading, not compiled, not run on hardware
//
#include <hip/hip_runtime.h>

typedef __bf16 bf16;
typedef __attribute__((ext_vector_type(16))) __bf16 v16bf;
typedef __attribute__((ext_vector_type(8)))  float  v8f;

// ---- TDM availability / arity detection (probe-verified heuristic) ----
#if defined(__has_builtin)
#  if __has_builtin(__builtin_amdgcn_tensor_load_to_lds)
#    define HAVE_TDM 1
#  endif
#endif
#ifndef HAVE_TDM
#  define HAVE_TDM 0
#endif
#if __has_include(<hip/amd_detail/amd_gfx1250_TDM.h>)
#  define TDM_6ARG 1   // therock-10.0 headers -> clang-23 6-arg builtin
#else
#  define TDM_6ARG 0   // ROCm 7.2 clang-22 5-arg builtin
#endif

#if HAVE_TDM
typedef __attribute__((ext_vector_type(4))) unsigned int u32x4;
typedef __attribute__((ext_vector_type(8))) int          i32x8;
typedef __attribute__((ext_vector_type(4))) int          i32x4;

// Issue a TDM load of a 2D tile (tl0 x tl1 elements of 2 bytes, row stride s0
// elements) from global to LDS. flags carries data_size + pad config (group1 w0).
// tl1 == 0 -> 1D transfer.
__device__ __forceinline__ void tdm_load(unsigned lds_off, const void* gaddr,
                                         unsigned flags, unsigned td0, unsigned td1,
                                         unsigned tl0, unsigned tl1, unsigned s0) {
    unsigned long long ga = (unsigned long long)gaddr;
    u32x4 g0;
    g0[0] = 1u;                                            // count=1, user mode
    g0[1] = lds_off;                                       // lds_addr
    g0[2] = (unsigned)ga;                                  // global_addr[31:0]
    g0[3] = (unsigned)((ga >> 32) & 0x01FFFFFFu) | 0x80000000u; // addr[56:32] | type=2
    i32x8 g1;
    g1[0] = (int)flags;                                    // wg_mask=0, data_size, pad cfg
    g1[1] = (int)(td0 << 16);                              // tensor_dim0[15:0]
    g1[2] = (int)((td0 >> 16) | (td1 << 16));              // tensor_dim0[31:16] | tensor_dim1[15:0]
    g1[3] = (int)((td1 >> 16) | (tl0 << 16));              // tensor_dim1[31:16] | tile_dim0
    g1[4] = (int)tl1;                                      // tile_dim1 | tile_dim2<<16
    g1[5] = (int)s0;                                       // tensor_dim0_stride[31:0]
    g1[6] = 0; g1[7] = 0;
    i32x4 z4 = {0, 0, 0, 0};
#if TDM_6ARG
    i32x8 z8 = {0, 0, 0, 0, 0, 0, 0, 0};
    __builtin_amdgcn_tensor_load_to_lds(g0, g1, z4, z4, z8, 0);
#else
    __builtin_amdgcn_tensor_load_to_lds(g0, g1, z4, z4, 0);
#endif
}

__device__ __forceinline__ void tdm_wait() {
#if defined(__has_builtin) && __has_builtin(__builtin_amdgcn_s_wait_tensorcnt)
    __builtin_amdgcn_s_wait_tensorcnt(0);
#else
    asm volatile("s_wait_tensorcnt 0x0" ::: "memory");
#endif
}
#endif // HAVE_TDM

__device__ __forceinline__ float leaky(float x) { return x > 0.f ? x : 0.1f * x; }

__device__ __forceinline__ unsigned pack_bf2(float a, float b) {
    union { __bf16 h[2]; unsigned u; } r;
    r.h[0] = (__bf16)a; r.h[1] = (__bf16)b;
    return r.u;
}

// CDNA5 16-bit A-matrix (16x32) lane layout: lane L holds M=L%16; element e -> K index
__device__ __forceinline__ int a_kidx(int lane, int e) {
    int v = e >> 1, h = e & 1;
    return ((lane & 16) ? 8 : 0) + ((v & 4) ? 16 : 0) + 2 * (v & 3) + h;
}
// CDNA5 16-bit B-matrix (32x16) lane layout: lane L holds N=L%16; element e -> K index
__device__ __forceinline__ int b_kidx(int lane, int e) {
    int v = e >> 1, h = e & 1;
    return ((lane & 16) ? 8 : 0) + ((v & 4) ? 16 : 0) + 2 * (v & 3) + h;
}

// Load an A fragment straight from a row-major [M][K] bf16 matrix using
// packed u32 pair loads (elements 2p,2p+1 sit at consecutive K).
__device__ __forceinline__ v16bf load_a_global(const bf16* w, int m, int Kdim, int kb, int lane) {
    union { v16bf v; unsigned u[8]; } r;
    const unsigned* wu = (const unsigned*)(w + (size_t)m * Kdim + kb);
    #pragma unroll
    for (int p = 0; p < 8; ++p) r.u[p] = wu[a_kidx(lane, 2 * p) >> 1];
    return r.v;
}

// ---------------- repack kernels ----------------

// NCHW fp32 -> NHWC bf16 via 32x32 LDS tile transpose (coalesced both sides)
__global__ void nchw_to_nhwc(const float* __restrict__ in, bf16* __restrict__ out,
                             int C, int HW) {
    __shared__ bf16 tile[32][33];
    const int hw0 = blockIdx.x * 32, c0 = blockIdx.y * 32, b = blockIdx.z;
    const int tx = threadIdx.x & 31, ty = threadIdx.x >> 5;   // 32 x 8
    const float* src = in + ((size_t)b * C + c0) * HW + hw0;
    for (int r = ty; r < 32; r += 8)                  // r: channel offset
        tile[r][tx] = (bf16)src[(size_t)r * HW + tx]; // read coalesced in hw
    __syncthreads();
    bf16* dst = out + ((size_t)b * HW + hw0) * C + c0;
    for (int r = ty; r < 32; r += 8)                  // r: pixel offset
        dst[(size_t)r * C + tx] = tile[tx][r];        // write coalesced in c
}

// (cout,cin,3,3) fp32 -> [tap][cout][cin] bf16
__global__ void pack_w3(const float* __restrict__ w, bf16* __restrict__ wp, int Cout, int Cin) {
    int i = blockIdx.x * 256 + threadIdx.x;
    int total = 9 * Cout * Cin;
    if (i >= total) return;
    int ci = i % Cin; int t = i / Cin;
    int co = t % Cout; int tap = t / Cout;
    wp[i] = (bf16)w[((size_t)co * Cin + ci) * 9 + tap];
}

// (cout,cin,1,1) fp32 -> [cout][cin] bf16
__global__ void pack_w1(const float* __restrict__ w, bf16* __restrict__ wp, int n) {
    int i = blockIdx.x * 256 + threadIdx.x;
    if (i >= n) return;
    wp[i] = (bf16)w[i];
}

// ---------------- 3x3 conv as implicit GEMM on WMMA ----------------
// in: NHWC bf16 (B,H,W,Cin)  wp: [9][Cout][Cin] bf16  out: NHWC bf16 (B,H,W,Cout)
// 256 threads (8 waves) -> tile 64 pixels x 32 couts; weight tile via TDM DMA.
__global__ void conv3x3_wmma(const bf16* __restrict__ in, const bf16* __restrict__ wp,
                             const float* __restrict__ bias, bf16* __restrict__ out,
                             int H, int W, int Cin, int Cout) {
    __shared__ bf16 Bs[64][34];   // [pixel][channel] (+1 dword pad)
    __shared__ bf16 As[32][34];   // [cout][channel]  (+1 dword pad)
    const int HW   = H * W;
    const int tid  = threadIdx.x;
    const int lane = tid & 31;
    const int wave = tid >> 5;
    const int waveCT = wave & 1;
    const int wavePT = wave >> 1;
    const size_t pixBase  = (size_t)blockIdx.x * 64;
    const int    coutBase = blockIdx.y * 32;

    // staging coords for this thread (hoisted out of the K loop)
    const int sp  = tid >> 2;            // staged pixel 0..63
    const int sc0 = (tid & 3) * 8;       // staged channel offset
    const size_t spix = pixBase + sp;
    const int pb  = (int)(spix / HW);
    const int phw = (int)(spix % HW);
    const int ph = phw / W, pw = phw % W;
    const bf16* pixSrc = in + (((size_t)pb * HW + (size_t)ph * W + pw) * Cin) + sc0;

    v8f acc = {};

    for (int tap = 0; tap < 9; ++tap) {
        const int kh = tap / 3 - 1, kw = tap % 3 - 1;
        const int hs = ph + kh, ws = pw + kw;
        const bool inb = (unsigned)hs < (unsigned)H && (unsigned)ws < (unsigned)W;
        const bf16* tapSrc = pixSrc + (ptrdiff_t)(kh * W + kw) * Cin;
        const bf16* wSrc = wp + ((size_t)tap * Cout + coutBase) * Cin;

        for (int cb = 0; cb < Cin; cb += 32) {
            // stage activation tile (zero outside image)
            if (inb) {
                const bf16* src = tapSrc + cb;
                __builtin_prefetch(src + 32, 0, 3);
                #pragma unroll
                for (int j = 0; j < 8; ++j) Bs[sp][sc0 + j] = src[j];
            } else {
                #pragma unroll
                for (int j = 0; j < 8; ++j) Bs[sp][sc0 + j] = (bf16)0.f;
            }
#if HAVE_TDM
            // weight tile 32x32 (row stride Cin) DMA'd by wave 0; pad cfg inserts
            // 1 dword every 16 dwords -> matches As[32][34] layout.
            if (wave == 0) {
                tdm_load((unsigned)(size_t)&As[0][0], wSrc + cb,
                         0x10000u | (1u << 20) | (3u << 22),   // 2B elems, pad on, 16dw interval
                         32u, 32u, 32u, 32u, (unsigned)Cin);
                tdm_wait();
            }
#else
            {
                int co = tid >> 3;
                int c0 = (tid & 7) * 4;
                const bf16* src = wSrc + (size_t)co * Cin + cb + c0;
                #pragma unroll
                for (int j = 0; j < 4; ++j) As[co][c0 + j] = src[j];
            }
#endif
            __syncthreads();
            v16bf afrag, bfrag;
            #pragma unroll
            for (int e = 0; e < 16; ++e) {
                afrag[e] = As[waveCT * 16 + (lane & 15)][a_kidx(lane, e)];
                bfrag[e] = Bs[wavePT * 16 + (lane & 15)][b_kidx(lane, e)];
            }
            acc = __builtin_amdgcn_wmma_f32_16x16x32_bf16(
                false, afrag, false, bfrag, (short)0, acc, false, false);
            __syncthreads();
        }
    }
    // bias + leaky, packed bf16x2 NHWC stores
    #pragma unroll
    for (int r = 0; r < 8; r += 2) {
        int co = coutBase + waveCT * 16 + r + ((lane & 16) ? 8 : 0);
        size_t pix = pixBase + wavePT * 16 + (lane & 15);
        unsigned pk = pack_bf2(leaky(acc[r] + bias[co]), leaky(acc[r + 1] + bias[co + 1]));
        *(unsigned*)(out + pix * Cout + co) = pk;
    }
}

// ---------------- 3x3 box mean/std (reflect pad, unbiased var over 9) ----------------
__global__ void box_stats(const bf16* __restrict__ x, bf16* __restrict__ lmean,
                          bf16* __restrict__ lstd, int H, int W, int C, size_t total) {
    size_t idx = (size_t)blockIdx.x * 256 + threadIdx.x;
    if (idx >= total) return;
    int c = (int)(idx % C);
    size_t pix = idx / C;
    int w = (int)(pix % W); size_t t = pix / W;
    int h = (int)(t % H);   int b = (int)(t / H);
    float s = 0.f, sq = 0.f;
    #pragma unroll
    for (int dh = -1; dh <= 1; ++dh) {
        int hh = h + dh; hh = hh < 0 ? 1 : (hh >= H ? H - 2 : hh);
        #pragma unroll
        for (int dw = -1; dw <= 1; ++dw) {
            int ww = w + dw; ww = ww < 0 ? 1 : (ww >= W ? W - 2 : ww);
            float v = (float)x[(((size_t)b * H + hh) * W + ww) * C + c];
            s += v; sq += v * v;
        }
    }
    float m   = s * (1.f / 9.f);
    float var = (sq - s * s * (1.f / 9.f)) * (1.f / 8.f) + 1e-5f;
    lmean[idx] = (bf16)m;
    lstd[idx]  = (bf16)sqrtf(var);
}

// ---------------- fused two-stage 1x1 head on WMMA ----------------
// code: NHWC bf16 (pix, MID); w1:[MID][MID]; w2:[Cout][MID]; out: NCHW fp32
template <int MID>
__global__ void head_wmma(const bf16* __restrict__ code,
                          const bf16* __restrict__ w1, const float* __restrict__ b1,
                          const bf16* __restrict__ w2, const float* __restrict__ b2,
                          float* __restrict__ out, int H, int W, int Cout) {
    __shared__ bf16 X[64][MID + 2];
    __shared__ bf16 T[64][MID + 2];
    const int HW = H * W;
    const int tid = threadIdx.x, lane = tid & 31, wave = tid >> 5;
    const size_t pixBase = (size_t)blockIdx.x * 64;

#if HAVE_TDM
    // 64 x MID contiguous tile, 1D DMA with pad: +1 dword after every MID/2 dwords
    // -> matches X[64][MID+2] layout. interval code: 2<<c dwords.
    if (wave == 0) {
        constexpr unsigned LEN = 64u * (unsigned)MID;
        constexpr unsigned IC  = (MID == 32) ? 3u : (MID == 64) ? 4u : (MID == 128) ? 5u : 6u;
        tdm_load((unsigned)(size_t)&X[0][0], code + pixBase * MID,
                 0x10000u | (1u << 20) | (IC << 22), LEN, 1u, LEN, 0u, LEN);
        tdm_wait();
    }
#else
    for (int i = tid; i < 64 * (MID / 4); i += 256) {
        int p = i / (MID / 4), c0 = (i % (MID / 4)) * 4;
        const bf16* src = code + ((pixBase + p) * MID + c0);
        #pragma unroll
        for (int j = 0; j < 4; ++j) X[p][c0 + j] = src[j];
    }
#endif
    __syncthreads();

    // GEMM1: T = leaky(W1 * X + b1)  (MID x 64)
    const int nT1 = (MID / 16) * 4;
    for (int t = wave; t < nT1; t += 8) {
        int ct = t / 4, pt = t % 4;
        v8f acc = {};
        for (int kb = 0; kb < MID; kb += 32) {
            v16bf a = load_a_global(w1, ct * 16 + (lane & 15), MID, kb, lane);
            v16bf bb;
            #pragma unroll
            for (int e = 0; e < 16; ++e)
                bb[e] = X[pt * 16 + (lane & 15)][kb + b_kidx(lane, e)];
            acc = __builtin_amdgcn_wmma_f32_16x16x32_bf16(false, a, false, bb, (short)0, acc, false, false);
        }
        #pragma unroll
        for (int r = 0; r < 8; r += 2) {
            int co = ct * 16 + r + ((lane & 16) ? 8 : 0);
            int p  = pt * 16 + (lane & 15);
            *(unsigned*)&T[p][co] =
                pack_bf2(leaky(acc[r] + b1[co]), leaky(acc[r + 1] + b1[co + 1]));
        }
    }
    __syncthreads();

    // GEMM2: out = W2 * T + b2  (Cout x 64), fp32 NCHW stores
    const int nT2 = (Cout / 16) * 4;
    for (int t = wave; t < nT2; t += 8) {
        int ct = t / 4, pt = t % 4;
        v8f acc = {};
        for (int kb = 0; kb < MID; kb += 32) {
            v16bf a = load_a_global(w2, ct * 16 + (lane & 15), MID, kb, lane);
            v16bf bb;
            #pragma unroll
            for (int e = 0; e < 16; ++e)
                bb[e] = T[pt * 16 + (lane & 15)][kb + b_kidx(lane, e)];
            acc = __builtin_amdgcn_wmma_f32_16x16x32_bf16(false, a, false, bb, (short)0, acc, false, false);
        }
        #pragma unroll
        for (int r = 0; r < 8; ++r) {
            int co = ct * 16 + r + ((lane & 16) ? 8 : 0);
            size_t pix = pixBase + pt * 16 + (lane & 15);
            int b  = (int)(pix / HW);
            int hw = (int)(pix % HW);
            out[((size_t)b * Cout + co) * HW + hw] = acc[r] + b2[co];
        }
    }
}

// ---------------- global mean/std + broadcast fill ----------------
__global__ void global_stats(const float* __restrict__ ori, float* __restrict__ gmean,
                             float* __restrict__ gstd, int HW) {
    const int bc = blockIdx.x;
    const float* src = ori + (size_t)bc * HW;
    __shared__ float ssum[256], ssq[256];
    float s = 0.f, sq = 0.f;
    for (int i = threadIdx.x; i < HW; i += 256) { float v = src[i]; s += v; sq += v * v; }
    ssum[threadIdx.x] = s; ssq[threadIdx.x] = sq;
    __syncthreads();
    for (int off = 128; off; off >>= 1) {
        if ((int)threadIdx.x < off) {
            ssum[threadIdx.x] += ssum[threadIdx.x + off];
            ssq[threadIdx.x]  += ssq[threadIdx.x + off];
        }
        __syncthreads();
    }
    float mean = ssum[0] / HW;
    float var  = (ssq[0] - ssum[0] * ssum[0] / HW) / (HW - 1) + 1e-5f;
    float sd   = sqrtf(var);
    float* gm = gmean + (size_t)bc * HW;
    float* gs = gstd  + (size_t)bc * HW;
    for (int i = threadIdx.x; i < HW; i += 256) { gm[i] = mean; gs[i] = sd; }
}

// ---------------- host ----------------

static void launch_head(int l, const bf16* code, const bf16* w1, const float* b1,
                        const bf16* w2, const float* b2, float* o,
                        int H, int W, int Cout, size_t totalPix, hipStream_t s) {
    dim3 g((unsigned)(totalPix / 64));
    switch (l) {
        case 0:  head_wmma<32><<<g, 256, 0, s>>>(code, w1, b1, w2, b2, o, H, W, Cout); break;
        case 1:  head_wmma<64><<<g, 256, 0, s>>>(code, w1, b1, w2, b2, o, H, W, Cout); break;
        case 2:  head_wmma<128><<<g, 256, 0, s>>>(code, w1, b1, w2, b2, o, H, W, Cout); break;
        default: head_wmma<256><<<g, 256, 0, s>>>(code, w1, b1, w2, b2, o, H, W, Cout); break;
    }
}

extern "C" void kernel_launch(void* const* d_in, const int* in_sizes, int n_in,
                              void* d_out, int out_size, void* d_ws, size_t ws_size,
                              hipStream_t stream) {
    (void)in_sizes; (void)n_in; (void)out_size; (void)ws_size;
    static const int INC[4]  = {64, 128, 256, 512};
    static const int MIDC[4] = {32, 64, 128, 256};
    static const int HS[4]   = {256, 128, 64, 32};
    const int Bn = 8;

    float* out = (float*)d_out;
    size_t Nl[4], G = 0;
    for (int l = 0; l < 4; ++l) { Nl[l] = (size_t)Bn * INC[l] * HS[l] * HS[l]; G += Nl[l]; }
    size_t base[4]; { size_t a = 0; for (int l = 0; l < 4; ++l) { base[l] = a; a += Nl[l]; } }

    // jax pytree flatten of params dict: keys sorted -> mean(16), pre(16), std(16); b before w, c1 before c2
    auto P = [&](int blk, int lvl, int conv, int wb) -> const float* {
        return (const float*)d_in[4 + blk * 16 + lvl * 4 + conv * 2 + wb];
    };

    for (int l = 0; l < 4; ++l) {
        const int C = INC[l], M = MIDC[l], H = HS[l], W = HS[l];
        const int HW = H * W;
        const size_t totalPix = (size_t)Bn * HW;

        size_t off = 0;
        auto alloc = [&](size_t bytes) -> void* {
            off = (off + 255) & ~(size_t)255;
            void* p = (char*)d_ws + off;
            off += bytes;
            return p;
        };
        bf16* nhwc = (bf16*)alloc(totalPix * C * 2);
        bf16* x1   = (bf16*)alloc(totalPix * M * 2);
        bf16* x2   = (bf16*)alloc(totalPix * M * 2);
        bf16* lm   = (bf16*)alloc(totalPix * M * 2);
        bf16* ls   = (bf16*)alloc(totalPix * M * 2);
        bf16* wc1  = (bf16*)alloc((size_t)9 * M * C * 2);
        bf16* wc2  = (bf16*)alloc((size_t)9 * M * M * 2);
        bf16* wm1  = (bf16*)alloc((size_t)M * M * 2);
        bf16* wm2  = (bf16*)alloc((size_t)C * M * 2);
        bf16* ws1  = (bf16*)alloc((size_t)M * M * 2);
        bf16* ws2  = (bf16*)alloc((size_t)C * M * 2);

        const float* feat = (const float*)d_in[l];

        nchw_to_nhwc<<<dim3(HW / 32, C / 32, Bn), 256, 0, stream>>>(feat, nhwc, C, HW);
        pack_w3<<<dim3((9 * M * C + 255) / 256), 256, 0, stream>>>(P(1, l, 0, 1), wc1, M, C);
        pack_w3<<<dim3((9 * M * M + 255) / 256), 256, 0, stream>>>(P(1, l, 1, 1), wc2, M, M);
        pack_w1<<<dim3((M * M + 255) / 256), 256, 0, stream>>>(P(0, l, 0, 1), wm1, M * M);
        pack_w1<<<dim3((C * M + 255) / 256), 256, 0, stream>>>(P(0, l, 1, 1), wm2, C * M);
        pack_w1<<<dim3((M * M + 255) / 256), 256, 0, stream>>>(P(2, l, 0, 1), ws1, M * M);
        pack_w1<<<dim3((C * M + 255) / 256), 256, 0, stream>>>(P(2, l, 1, 1), ws2, C * M);

        conv3x3_wmma<<<dim3((unsigned)(totalPix / 64), M / 32), 256, 0, stream>>>(
            nhwc, wc1, P(1, l, 0, 0), x1, H, W, C, M);
        conv3x3_wmma<<<dim3((unsigned)(totalPix / 64), M / 32), 256, 0, stream>>>(
            x1, wc2, P(1, l, 1, 0), x2, H, W, M, M);
        box_stats<<<dim3((unsigned)((totalPix * M + 255) / 256)), 256, 0, stream>>>(
            x2, lm, ls, H, W, M, totalPix * M);
        launch_head(l, lm, wm1, P(0, l, 0, 0), wm2, P(0, l, 1, 0), out + base[l], H, W, C, totalPix, stream);
        launch_head(l, ls, ws1, P(2, l, 0, 0), ws2, P(2, l, 1, 0), out + G + base[l], H, W, C, totalPix, stream);
        global_stats<<<dim3(Bn * C), 256, 0, stream>>>(feat, out + 2 * G + base[l], out + 3 * G + base[l], HW);
    }
}
